// GraphNeuralNetwork_75222057222370
// MI455X (gfx1250) — matrix operand order repaired and checked by reading, not compiled, not run on hardware
//
#include <hip/hip_runtime.h>
#include <hip/hip_bf16.h>
#include <math.h>

// Problem constants (match reference)
#define BB        32
#define F_INS     78
#define F_PAD     96      // F_IN zero-padded up to a multiple of 32
#define HIDS      128
#define HEADSN    4
#define LAYERSN   4
#define HEAD_DIMS 32
#define MEM_LENS  64
#define NEG_SLOPEF 0.2f

typedef __attribute__((ext_vector_type(16))) _Float16 v16h_t;
typedef __attribute__((ext_vector_type(8)))  float    v8f_t;

// ---------------------------------------------------------------------------
// Fragment builders (fp32 memory -> fp16 WMMA fragments, ISA 7.12.2 layouts)
// ---------------------------------------------------------------------------
__device__ __forceinline__ v16h_t cvt16(float4 x0, float4 x1, float4 x2, float4 x3) {
  v16h_t r;
  r[0]  = (_Float16)x0.x; r[1]  = (_Float16)x0.y; r[2]  = (_Float16)x0.z; r[3]  = (_Float16)x0.w;
  r[4]  = (_Float16)x1.x; r[5]  = (_Float16)x1.y; r[6]  = (_Float16)x1.z; r[7]  = (_Float16)x1.w;
  r[8]  = (_Float16)x2.x; r[9]  = (_Float16)x2.y; r[10] = (_Float16)x2.z; r[11] = (_Float16)x2.w;
  r[12] = (_Float16)x3.x; r[13] = (_Float16)x3.y; r[14] = (_Float16)x3.z; r[15] = (_Float16)x3.w;
  return r;
}
// A 16x32 fragment: lane holds A[row][k0+half*8 + 0..7], A[row][k0+16+half*8 + 0..7]
// (pass p = Arow_ptr + k0 + half*8)
__device__ __forceinline__ v16h_t frag_a(const float* __restrict__ p) {
  return cvt16(*(const float4*)(p),      *(const float4*)(p + 4),
               *(const float4*)(p + 16), *(const float4*)(p + 20));
}
// B 32x16 fragment: lane holds B[col][k0+half*8 + 0..7], B[col][k0+16+half*8 + 0..7]
// (pass p = Bcol_ptr + k0 + half*8)
__device__ __forceinline__ v16h_t frag_b(const float* __restrict__ p) {
  return cvt16(*(const float4*)(p),      *(const float4*)(p + 4),
               *(const float4*)(p + 16), *(const float4*)(p + 20));
}

// ---------------------------------------------------------------------------
// Fast WMMA GEMM:  C[M x N] = A[M x K] @ B[N x K]^T (+bias[N]) (+relu)
// REQUIRES: M % 32 == 0, N % 32 == 0, K % 32 == 0 (call sites guarantee this).
// One wave computes a 32x32 macro-tile (2x2 of 16x16 WMMA tiles) so each A/B
// fragment is reused twice: 16x global_load_b128 feed 4x v_wmma per K-step.
// Branch-free inner loop: EXEC stays all-ones (WMMA requirement).
// ---------------------------------------------------------------------------
__global__ void __launch_bounds__(128)
wmma_gemm_kernel(const float* __restrict__ A,
                 const float* __restrict__ Bm,
                 const float* __restrict__ bias,
                 float* __restrict__ C,
                 int M, int N, int K, int relu) {
  const int tn  = blockIdx.x;                      // 32-col macro tile
  const int tmw = blockIdx.y * 4 + threadIdx.y;    // 32-row macro tile (wave-uniform)
  if (tmw * 32 >= M) return;                       // uniform per wave

  const int lane = threadIdx.x;                    // 0..31
  const int half = lane >> 4;                      // lane group 0/1
  const int idx  = lane & 15;                      // row (A) / col (B) in tile

  const float* __restrict__ Ap0 = A  + (size_t)(tmw * 32 + idx) * K       + half * 8;
  const float* __restrict__ Ap1 = A  + (size_t)(tmw * 32 + 16 + idx) * K  + half * 8;
  const float* __restrict__ Bp0 = Bm + (size_t)(tn * 32 + idx) * K        + half * 8;
  const float* __restrict__ Bp1 = Bm + (size_t)(tn * 32 + 16 + idx) * K   + half * 8;

  v8f_t acc00 = {0.f,0.f,0.f,0.f,0.f,0.f,0.f,0.f};
  v8f_t acc01 = acc00, acc10 = acc00, acc11 = acc00;

  for (int k0 = 0; k0 < K; k0 += 32) {
    const v16h_t a0 = frag_a(Ap0 + k0);
    const v16h_t a1 = frag_a(Ap1 + k0);
    const v16h_t b0 = frag_b(Bp0 + k0);
    const v16h_t b1 = frag_b(Bp1 + k0);
    acc00 = __builtin_amdgcn_wmma_f32_16x16x32_f16(false, a0, false, b0, (short)0, acc00, false, false);
    acc01 = __builtin_amdgcn_wmma_f32_16x16x32_f16(false, a0, false, b1, (short)0, acc01, false, false);
    acc10 = __builtin_amdgcn_wmma_f32_16x16x32_f16(false, a1, false, b0, (short)0, acc10, false, false);
    acc11 = __builtin_amdgcn_wmma_f32_16x16x32_f16(false, a1, false, b1, (short)0, acc11, false, false);
  }

  // C/D layout: VGPR r -> M = r + 8*half, N = idx (per 16x16 sub-tile)
  const int c0 = tn * 32 + idx;
  const int c1 = c0 + 16;
  const float bv0 = bias ? bias[c0] : 0.f;
  const float bv1 = bias ? bias[c1] : 0.f;
#pragma unroll
  for (int r = 0; r < 8; ++r) {
    const int r0 = tmw * 32 + r + half * 8;
    const int r1 = r0 + 16;
    float v00 = acc00[r] + bv0, v01 = acc01[r] + bv1;
    float v10 = acc10[r] + bv0, v11 = acc11[r] + bv1;
    if (relu) {
      v00 = fmaxf(v00, 0.f); v01 = fmaxf(v01, 0.f);
      v10 = fmaxf(v10, 0.f); v11 = fmaxf(v11, 0.f);
    }
    C[(size_t)r0 * N + c0] = v00;
    C[(size_t)r0 * N + c1] = v01;
    C[(size_t)r1 * N + c0] = v10;
    C[(size_t)r1 * N + c1] = v11;
  }
}

// ---------------------------------------------------------------------------
// Small helpers
// ---------------------------------------------------------------------------
__global__ void fill_kernel(float* __restrict__ p, float v, int n) {
  int i = blockIdx.x * blockDim.x + threadIdx.x;
  if (i < n) p[i] = v;
}

// dst[N x Kdst] = src[N x Ksrc] zero-padded along K
__global__ void pad_rows_kernel(const float* __restrict__ src,
                                float* __restrict__ dst,
                                int N, int Ksrc, int Kdst) {
  int i = blockIdx.x * blockDim.x + threadIdx.x;
  if (i >= N * Kdst) return;
  int r = i / Kdst, c = i - r * Kdst;
  dst[i] = (c < Ksrc) ? src[(size_t)r * Ksrc + c] : 0.f;
}

__device__ inline void atomicMaxFloat(float* addr, float val) {
  int* ia  = (int*)addr;
  int  old = *ia;
  while (__int_as_float(old) < val) {
    int assumed = old;
    old = atomicCAS(ia, assumed, __float_as_int(val));
    if (old == assumed) break;
  }
}

// Per-(node,head) attention coefficients: asrc = <h_nh, a_src_h>, adst likewise
__global__ void att_coeff_kernel(const float* __restrict__ hh,
                                 const float* __restrict__ a_src,
                                 const float* __restrict__ a_dst,
                                 float* __restrict__ asrc,
                                 float* __restrict__ adst, int N) {
  int i = blockIdx.x * blockDim.x + threadIdx.x;
  if (i >= N * HEADSN) return;
  int n = i / HEADSN, h = i - n * HEADSN;
  const float* hp = hh + (size_t)n * HEADSN * HIDS + h * HIDS;
  float s1 = 0.f, s2 = 0.f;
  for (int c = 0; c < HIDS; ++c) {
    float v = hp[c];
    s1 += v * a_src[h * HIDS + c];
    s2 += v * a_dst[h * HIDS + c];
  }
  asrc[i] = s1;
  adst[i] = s2;
}

// Edge logits + segment max (edges [0,E0) are real, [E0, E0+N) are self loops)
__global__ void edge_logits_kernel(const int* __restrict__ ei, int E0, int N,
                                   const float* __restrict__ asrc,
                                   const float* __restrict__ adst,
                                   float* __restrict__ ebuf,
                                   float* __restrict__ m) {
  int i = blockIdx.x * blockDim.x + threadIdx.x;
  int tot = (E0 + N) * HEADSN;
  if (i >= tot) return;
  int ed = i / HEADSN, h = i - ed * HEADSN;
  int s = (ed < E0) ? ei[ed]      : (ed - E0);
  int d = (ed < E0) ? ei[E0 + ed] : (ed - E0);
  float x = asrc[s * HEADSN + h] + adst[d * HEADSN + h];
  float e = (x > 0.f) ? x : NEG_SLOPEF * x;
  ebuf[i] = e;
  atomicMaxFloat(&m[d * HEADSN + h], e);
}

// ex = exp(e - m[dst]); segment sum
__global__ void edge_exp_kernel(const int* __restrict__ ei, int E0, int N,
                                const float* __restrict__ m,
                                float* __restrict__ ebuf,
                                float* __restrict__ ssum) {
  int i = blockIdx.x * blockDim.x + threadIdx.x;
  int tot = (E0 + N) * HEADSN;
  if (i >= tot) return;
  int ed = i / HEADSN, h = i - ed * HEADSN;
  int d = (ed < E0) ? ei[E0 + ed] : (ed - E0);
  float ex = __expf(ebuf[i] - m[d * HEADSN + h]);
  ebuf[i] = ex;
  atomicAdd(&ssum[d * HEADSN + h], ex);
}

// Weighted message aggregation: one wave per (edge, head), lane strides channels
__global__ void edge_aggregate_kernel(const int* __restrict__ ei, int E0, int N,
                                      const float* __restrict__ hh,
                                      const float* __restrict__ ex,
                                      const float* __restrict__ ssum,
                                      float* __restrict__ agg) {
  int gid  = blockIdx.x * blockDim.x + threadIdx.x;
  int wid  = gid >> 5;
  int lane = gid & 31;
  int tot  = (E0 + N) * HEADSN;
  if (wid >= tot) return;
  int ed = wid / HEADSN, h = wid - ed * HEADSN;
  int s = (ed < E0) ? ei[ed]      : (ed - E0);
  int d = (ed < E0) ? ei[E0 + ed] : (ed - E0);
  float alpha = ex[wid] / ssum[d * HEADSN + h];
  const float* hp = hh  + (size_t)s * HEADSN * HIDS + h * HIDS;
  float*       ap = agg + (size_t)d * HEADSN * HIDS + h * HIDS;
#pragma unroll
  for (int c = 0; c < HIDS / 32; ++c)
    atomicAdd(&ap[lane + 32 * c], hp[lane + 32 * c] * alpha);
}

// mean over heads + bias + relu
__global__ void head_mean_kernel(const float* __restrict__ agg,
                                 const float* __restrict__ b,
                                 float* __restrict__ hout, int N) {
  int i = blockIdx.x * blockDim.x + threadIdx.x;
  if (i >= N * HIDS) return;
  int n = i / HIDS, c = i - n * HIDS;
  float s = 0.f;
#pragma unroll
  for (int h = 0; h < HEADSN; ++h)
    s += agg[(size_t)n * HEADSN * HIDS + h * HIDS + c];
  hout[i] = fmaxf(s * (1.f / HEADSN) + b[c], 0.f);
}

// graph_id is repeat(arange(B), nper) -> contiguous segments, no atomics
__global__ void pool_mean_kernel(const float* __restrict__ h,
                                 float* __restrict__ pooled, int nper) {
  int g = blockIdx.x, c = threadIdx.x;   // blockDim = HIDS
  const float* p = h + (size_t)g * nper * HIDS + c;
  float s = 0.f;
  for (int n = 0; n < nper; ++n) s += p[(size_t)n * HIDS];
  pooled[g * HIDS + c] = s / (float)nper;
}

// Tiny MHA: one block of 64 threads per (batch, head)
__global__ void mha_attn_kernel(const float* __restrict__ q,
                                const float* __restrict__ k,
                                const float* __restrict__ v,
                                float* __restrict__ o) {
  int b = blockIdx.x, h = blockIdx.y, t = threadIdx.x;  // t in [0,64)
  __shared__ float sc[MEM_LENS];
  const float* qp = q + b * HIDS + h * HEAD_DIMS;
  const float* kp = k + ((size_t)b * MEM_LENS + t) * HIDS + h * HEAD_DIMS;
  float s = 0.f;
  for (int d = 0; d < HEAD_DIMS; ++d) s += qp[d] * kp[d];
  sc[t] = s * 0.17677669529663687f;  // 1/sqrt(32)
  __syncthreads();
  float mx = -1e30f;
  for (int i = 0; i < MEM_LENS; ++i) mx = fmaxf(mx, sc[i]);
  float ex = __expf(sc[t] - mx);
  __syncthreads();
  sc[t] = ex;
  __syncthreads();
  float sum = 0.f;
  for (int i = 0; i < MEM_LENS; ++i) sum += sc[i];
  if (t < HEAD_DIMS) {
    const float* vp = v + (size_t)b * MEM_LENS * HIDS + h * HEAD_DIMS + t;
    float acc = 0.f;
    for (int kk = 0; kk < MEM_LENS; ++kk) acc += sc[kk] * vp[(size_t)kk * HIDS];
    o[b * HIDS + h * HEAD_DIMS + t] = acc / sum;
  }
}

__global__ void concat_kernel(const float* __restrict__ a,
                              const float* __restrict__ b,
                              float* __restrict__ c, int n, int d) {
  int i = blockIdx.x * blockDim.x + threadIdx.x;
  if (i >= n * 2 * d) return;
  int g = i / (2 * d), cc = i - g * 2 * d;
  c[i] = (cc < d) ? a[g * d + cc] : b[g * d + (cc - d)];
}

__global__ void final_kernel(const float* __restrict__ h2,
                             const float* __restrict__ W3,
                             const float* __restrict__ b3,
                             float* __restrict__ out) {
  int g = threadIdx.x;
  if (g >= BB) return;
  float s = 0.f;
  for (int c = 0; c < HIDS; ++c) s += h2[g * HIDS + c] * W3[c];
  s += b3[0];
  out[g] = 1.f / (1.f + __expf(-s));
}

// ---------------------------------------------------------------------------
extern "C" void kernel_launch(void* const* d_in, const int* in_sizes, int n_in,
                              void* d_out, int out_size, void* d_ws, size_t ws_size,
                              hipStream_t stream) {
  const float* lig_x    = (const float*)d_in[0];
  const float* prot_x   = (const float*)d_in[1];
  const float* mem      = (const float*)d_in[2];
  const int*   lig_ei   = (const int*)d_in[3];
  const int*   prot_ei  = (const int*)d_in[4];
  // d_in[5], d_in[6] graph ids: segments are contiguous, handled analytically
  const float* W_emb    = (const float*)d_in[7];
  const float* b_emb    = (const float*)d_in[8];
  const float* gat_W    = (const float*)d_in[9];
  const float* gat_asrc = (const float*)d_in[10];
  const float* gat_adst = (const float*)d_in[11];
  const float* gat_b    = (const float*)d_in[12];
  const float* ipw      = (const float*)d_in[13];
  const float* ipb      = (const float*)d_in[14];
  const float* out_w    = (const float*)d_in[15];
  const float* out_b    = (const float*)d_in[16];
  const float* W1       = (const float*)d_in[17];
  const float* b1       = (const float*)d_in[18];
  const float* W2       = (const float*)d_in[19];
  const float* b2       = (const float*)d_in[20];
  const float* W3       = (const float*)d_in[21];
  const float* b3       = (const float*)d_in[22];

  const int N_lig  = in_sizes[0] / F_INS;   // 1536
  const int N_prot = in_sizes[1] / F_INS;   // 32768
  const int E_lig  = in_sizes[3] / 2;       // 6144
  const int E_prot = in_sizes[4] / 2;       // 262144
  const int N_MAX  = (N_prot > N_lig) ? N_prot : N_lig;
  const int ET_lig = E_lig + N_lig, ET_prot = E_prot + N_prot;
  const int ET_MAX = (ET_prot > ET_lig) ? ET_prot : ET_lig;

  // Bump allocator over workspace (all buffers re-initialized every call)
  char* wp = (char*)d_ws;
  auto alloc = [&](size_t elems) -> float* {
    float* r = (float*)wp;
    wp += ((elems * sizeof(float) + 255) & ~(size_t)255);
    return r;
  };
  float* hbuf    = alloc((size_t)N_MAX * HIDS);            // node features
  float* hhbuf   = alloc((size_t)N_MAX * HEADSN * HIDS);   // transformed (N x 512)
  float* aggbuf  = alloc((size_t)N_MAX * HEADSN * HIDS);   // message accumulators
  float* xpad    = alloc((size_t)N_MAX * F_PAD);           // K-padded node inputs
  float* wembpad = alloc((size_t)HIDS * F_PAD);            // K-padded W_emb
  float* asrcb   = alloc((size_t)N_MAX * HEADSN);
  float* adstb   = alloc((size_t)N_MAX * HEADSN);
  float* ebuf    = alloc((size_t)ET_MAX * HEADSN);
  float* mbuf    = alloc((size_t)N_MAX * HEADSN);
  float* sbuf    = alloc((size_t)N_MAX * HEADSN);
  float* pooled_lig  = alloc(BB * HIDS);
  float* pooled_prot = alloc(BB * HIDS);
  float* qb   = alloc(BB * HIDS);
  float* kb   = alloc((size_t)BB * MEM_LENS * HIDS);
  float* vb   = alloc((size_t)BB * MEM_LENS * HIDS);
  float* ab   = alloc(BB * HIDS);
  float* lf   = alloc(BB * HIDS);
  float* comb = alloc(BB * 2 * HIDS);
  float* h1b  = alloc(BB * 256);
  float* h2b  = alloc(BB * HIDS);

  // M % 32 == 0, N % 32 == 0, K % 32 == 0 required (guaranteed by callers)
  auto gemm = [&](const float* A, const float* Bm, const float* bias, float* C,
                  int M, int N, int K, int relu) {
    dim3 g(N / 32, (M + 127) / 128);
    dim3 blk(32, 4);
    wmma_gemm_kernel<<<g, blk, 0, stream>>>(A, Bm, bias, C, M, N, K, relu);
  };

  // Pad W_emb (128 x 78) -> (128 x 96) once
  pad_rows_kernel<<<(HIDS * F_PAD + 255) / 256, 256, 0, stream>>>(
      W_emb, wembpad, HIDS, F_INS, F_PAD);

  auto encode = [&](const float* x, int N, const int* ei, int E0, float* pooled) {
    // pad x (N x 78) -> (N x 96), then h = xpad @ wembpad^T + b_emb
    pad_rows_kernel<<<(N * F_PAD + 255) / 256, 256, 0, stream>>>(
        x, xpad, N, F_INS, F_PAD);
    gemm(xpad, wembpad, b_emb, hbuf, N, HIDS, F_PAD, 0);
    const int nt = (E0 + N) * HEADSN;
    for (int l = 0; l < LAYERSN; ++l) {
      gemm(hbuf, gat_W + (size_t)l * HEADSN * HIDS * HIDS, nullptr, hhbuf,
           N, HEADSN * HIDS, HIDS, 0);
      att_coeff_kernel<<<(N * HEADSN + 255) / 256, 256, 0, stream>>>(
          hhbuf, gat_asrc + (size_t)l * HEADSN * HIDS,
          gat_adst + (size_t)l * HEADSN * HIDS, asrcb, adstb, N);
      fill_kernel<<<(N * HEADSN + 255) / 256, 256, 0, stream>>>(mbuf, -1e30f, N * HEADSN);
      fill_kernel<<<(N * HEADSN + 255) / 256, 256, 0, stream>>>(sbuf, 0.f, N * HEADSN);
      fill_kernel<<<(N * HEADSN * HIDS + 255) / 256, 256, 0, stream>>>(
          aggbuf, 0.f, N * HEADSN * HIDS);
      edge_logits_kernel<<<(nt + 255) / 256, 256, 0, stream>>>(
          ei, E0, N, asrcb, adstb, ebuf, mbuf);
      edge_exp_kernel<<<(nt + 255) / 256, 256, 0, stream>>>(
          ei, E0, N, mbuf, ebuf, sbuf);
      edge_aggregate_kernel<<<(nt * 32 + 255) / 256, 256, 0, stream>>>(
          ei, E0, N, hhbuf, ebuf, sbuf, aggbuf);
      head_mean_kernel<<<(N * HIDS + 255) / 256, 256, 0, stream>>>(
          aggbuf, gat_b + (size_t)l * HIDS, hbuf, N);
    }
    pool_mean_kernel<<<BB, HIDS, 0, stream>>>(hbuf, pooled, N / BB);
  };

  encode(lig_x,  N_lig,  lig_ei,  E_lig,  pooled_lig);
  encode(prot_x, N_prot, prot_ei, E_prot, pooled_prot);

  // MHA (query = pooled ligand, memory = context)
  gemm(pooled_lig, ipw,                   ipb,            qb, BB,             HIDS, HIDS, 0);
  gemm(mem,        ipw + HIDS * HIDS,     ipb + HIDS,     kb, BB * MEM_LENS, HIDS, HIDS, 0);
  gemm(mem,        ipw + 2 * HIDS * HIDS, ipb + 2 * HIDS, vb, BB * MEM_LENS, HIDS, HIDS, 0);
  mha_attn_kernel<<<dim3(BB, HEADSN), MEM_LENS, 0, stream>>>(qb, kb, vb, ab);
  gemm(ab, out_w, out_b, lf, BB, HIDS, HIDS, 0);

  // MLP head
  concat_kernel<<<(BB * 2 * HIDS + 255) / 256, 256, 0, stream>>>(
      lf, pooled_prot, comb, BB, HIDS);
  gemm(comb, W1, b1, h1b, BB, 256, 2 * HIDS, 1);
  gemm(h1b,  W2, b2, h2b, BB, HIDS, 256,     1);
  final_kernel<<<1, 32, 0, stream>>>(h2b, W3, b3, (float*)d_out);
}
